// GCNWithAttention_sign_28157805592584
// MI455X (gfx1250) — matrix operand order repaired and checked by reading, not compiled, not run on hardware
//
#include <hip/hip_runtime.h>

#define NN 10000
#define KK 64
#define DD 128
#define HH 256

typedef __attribute__((ext_vector_type(16))) _Float16 v16h;
typedef __attribute__((ext_vector_type(8)))  float    v8f;

union Frag { v16h h; uint4 q[2]; };

static constexpr int AST = 264;   // f16 row stride, activation buffers (64 x 256)
static constexpr int WST = 264;   // f16 row stride, transposed weight buffer (256 x 256)

// ---------------------------------------------------------------------------
// Kernel 1: zero the whole output (Y_pred + 400MB pairwise) with 128-bit stores
// ---------------------------------------------------------------------------
__global__ void __launch_bounds__(256)
zero_out_kernel(float4* __restrict__ out, long long n4) {
  long long idx    = (long long)blockIdx.x * blockDim.x + threadIdx.x;
  long long stride = (long long)gridDim.x * blockDim.x;
  float4 z; z.x = z.y = z.z = z.w = 0.0f;
  for (long long i = idx; i < n4; i += stride) out[i] = z;
}

// ---------------------------------------------------------------------------
// One dense layer: dst(64x256) = relu(src(64x256) @ Wt^T + bias) in f16 WMMA.
// src: row-major 64xAST f16 LDS;  wt: transposed weights, wt[n*WST+k] = W[k][n].
// 8 waves: wave w owns n-tiles {2w, 2w+1}; each does 4 m-tiles x 8 k-steps.
// ---------------------------------------------------------------------------
__device__ __forceinline__ void mlp_layer(const _Float16* __restrict__ src,
                                          const _Float16* __restrict__ wt,
                                          _Float16* __restrict__ dst,
                                          const float* __restrict__ bias) {
  const int wave = threadIdx.x >> 5;
  const int lane = threadIdx.x & 31;
  const int half = lane >> 4;   // which 16-lane half
  const int lr   = lane & 15;

  for (int nt = 2 * wave; nt < 2 * wave + 2; ++nt) {
    const float bv = bias[nt * 16 + lr];   // per-lane column bias (N = nt*16+lr)
    for (int mt = 0; mt < 4; ++mt) {
      v8f acc = {};
#pragma unroll
      for (int kt = 0; kt < 8; ++kt) {
        Frag a, b;
        // A frag (16x32 f16): lane<16 -> K {0..7, 16..23}; lane>=16 -> K {8..15, 24..31}
        const int arow = mt * 16 + lr;
        const int acol = kt * 32 + half * 8;
        a.q[0] = *(const uint4*)&src[arow * AST + acol];
        a.q[1] = *(const uint4*)&src[arow * AST + acol + 16];
        // B frag (32x16 f16): lane<16 -> K {0..7, 16..23}; lane>=16 -> K {8..15, 24..31}
        const int brow = nt * 16 + lr;
        const int bcol = kt * 32 + half * 8;
        b.q[0] = *(const uint4*)&wt[brow * WST + bcol];
        b.q[1] = *(const uint4*)&wt[brow * WST + bcol + 16];
        acc = __builtin_amdgcn_wmma_f32_16x16x32_f16(false, a.h, false, b.h,
                                                     (short)0, acc, false, false);
      }
      // C/D layout: lane gives N; VGPR v -> M = mt*16 + half*8 + v
#pragma unroll
      for (int v = 0; v < 8; ++v) {
        float r = fmaxf(acc[v] + bv, 0.0f);
        dst[(mt * 16 + half * 8 + v) * AST + nt * 16 + lr] = (_Float16)r;
      }
    }
  }
}

// ---------------------------------------------------------------------------
// Kernel 2: per-node MLP + softmax attention + scatter. One block per node.
// ---------------------------------------------------------------------------
__global__ void __launch_bounds__(256)
gcn_attn_kernel(const float* __restrict__ x,
                const int*   __restrict__ nb,
                const float* __restrict__ treat,
                const float* __restrict__ ehat,
                const float* __restrict__ W1, const float* __restrict__ b1,
                const float* __restrict__ W2, const float* __restrict__ b2,
                const float* __restrict__ W3, const float* __restrict__ b3,
                const float* __restrict__ bsc,
                float* __restrict__ Y, float* __restrict__ P) {
  __shared__ __align__(16) _Float16 sA[64 * AST];    // zc, reused as h2
  __shared__ __align__(16) _Float16 sB[64 * AST];    // h1
  __shared__ __align__(16) _Float16 sW[HH * WST];    // W1^T then W2^T (f16)
  __shared__ float sMLP[KK];
  __shared__ float sRED[KK];

  const int tid = threadIdx.x;
  const int i   = blockIdx.x;
  const int cur = nb[i * (KK + 1)];

  // Stage zc: row j = [ x[cur] (128) | x[chosen_j] (128) ], f32 -> f16
  for (int e = tid; e < KK * 256; e += 256) {
    const int r = e >> 8, c = e & 255;
    const int srcRow = (c < DD) ? cur : nb[i * (KK + 1) + 1 + r];
    sA[r * AST + c] = (_Float16)x[srcRow * DD + (c & (DD - 1))];
  }
  // Stage W1^T: sW[n*WST+k] = W1[k*256+n] (coalesced global reads)
  for (int e = tid; e < 2 * DD * HH; e += 256) {
    const int k = e >> 8, n = e & 255;
    sW[n * WST + k] = (_Float16)W1[e];
  }
  // Prefetch W2 (256KB) into cache while layer 1 runs
  for (int l = tid; l < (HH * HH) / 16; l += 256)
    __builtin_prefetch(&W2[l * 16], 0, 0);
  __syncthreads();

  mlp_layer(sA, sW, sB, b1);        // h1 = relu(zc @ W1 + b1)
  __syncthreads();

  // Stage W2^T (layer-1 reads of sW finished at the barrier)
  for (int e = tid; e < HH * HH; e += 256) {
    const int k = e >> 8, n = e & 255;
    sW[n * WST + k] = (_Float16)W2[e];
  }
  __syncthreads();

  mlp_layer(sB, sW, sA, b2);        // h2 = relu(h1 @ W2 + b2), into sA
  __syncthreads();

  // Layer 3: mlp[j] = h2[j] . W3 + b3  (threads 0..63, W3 broadcast from L2)
  if (tid < KK) {
    float acc = b3[0];
    for (int k = 0; k < HH; ++k) acc += (float)sA[tid * AST + k] * W3[k];
    sMLP[tid] = acc;
  }
  __syncthreads();

  // softmax(b*|mlp|) over the 64 neighbors; w = mlp * score; scatter + reduce
  if (tid < KK) {
    const float bb = bsc[0];
    float mx = -1e30f;
    for (int j = 0; j < KK; ++j) mx = fmaxf(mx, bb * fabsf(sMLP[j]));
    float s = 0.0f;
    for (int j = 0; j < KK; ++j) s += __expf(bb * fabsf(sMLP[j]) - mx);
    const float ml = sMLP[tid];
    const float wj = ml * (__expf(bb * fabsf(ml) - mx) / s);
    const int   c  = nb[i * (KK + 1) + 1 + tid];
    sRED[tid] = (treat[c] - ehat[c]) * wj;
    P[(size_t)cur * NN + c] = wj;            // pairwise scatter (diag never hit)
  }
  __syncthreads();
  if (tid == 0) {
    float s = 0.0f;
    for (int j = 0; j < KK; ++j) s += sRED[j];
    Y[cur] = s;
  }
}

// ---------------------------------------------------------------------------
extern "C" void kernel_launch(void* const* d_in, const int* in_sizes, int n_in,
                              void* d_out, int out_size, void* d_ws, size_t ws_size,
                              hipStream_t stream) {
  const float* x     = (const float*)d_in[0];
  const int*   nb    = (const int*)  d_in[1];
  const float* treat = (const float*)d_in[2];
  const float* ehat  = (const float*)d_in[3];
  const float* W1    = (const float*)d_in[4];
  const float* b1    = (const float*)d_in[5];
  const float* W2    = (const float*)d_in[6];
  const float* b2    = (const float*)d_in[7];
  const float* W3    = (const float*)d_in[8];
  const float* b3    = (const float*)d_in[9];
  const float* bsc   = (const float*)d_in[10];

  float* Y = (float*)d_out;
  float* P = Y + NN;

  const long long n4 = ((long long)NN + (long long)NN * NN) / 4;  // divisible by 4
  zero_out_kernel<<<2048, 256, 0, stream>>>((float4*)d_out, n4);
  gcn_attn_kernel<<<NN, 256, 0, stream>>>(x, nb, treat, ehat,
                                          W1, b1, W2, b2, W3, b3, bsc, Y, P);
}
